// HM2D_83227876262508
// MI455X (gfx1250) — matrix operand order repaired and checked by reading, not compiled, not run on hardware
//
#include <hip/hip_runtime.h>
#include <hip/hip_bf16.h>

#define D_MODEL 96
#define D_INNER 192
#define D_STATE 16
#define DT_RANK 6
#define KDIR    4
#define B_      2
#define H_      64
#define W_      64
#define L_      4096            // H_*W_
#define T_      8192            // B_*L_
#define CPROJ   152             // KDIR*(DT_RANK+2*D_STATE) = 4*38
#define NCHUNK  32
#define LCH     128             // L_/NCHUNK
#define NT      4               // N tiles per wave in the WMMA GEMM

typedef __attribute__((ext_vector_type(16))) __bf16 v16bf;
typedef __attribute__((ext_vector_type(8)))  float  v8f;

// ---------- helpers ----------
__device__ __forceinline__ unsigned short f2bf(float f) {
  union { float f; unsigned int u; } c; c.f = f;
  unsigned int u = c.u;
  u += 0x7fffu + ((u >> 16) & 1u);          // round-to-nearest-even
  return (unsigned short)(u >> 16);
}
__device__ __forceinline__ float softplus_f(float x) {
  return x > 20.f ? x : log1pf(__expf(x));
}
__device__ __forceinline__ float silu_f(float x) {
  return x / (1.f + __expf(-x));
}
// direction k, step s -> row-major spatial position p
__device__ __forceinline__ int dir_pos(int k, int s) {
  int ss = (k & 2) ? (L_ - 1 - s) : s;
  if (k & 1) { int h = ss & (W_ - 1); int w = ss >> 6; return (h << 6) + w; }
  return ss;
}

// ---------- fp32 -> bf16 bulk convert (4 elems/thread) ----------
__global__ void k_cvt_bf16(const float* __restrict__ src, unsigned short* __restrict__ dst,
                           int n4) {
  int i = blockIdx.x * blockDim.x + threadIdx.x;
  if (i >= n4) return;
  float4 f = ((const float4*)src)[i];
  uint2 r;
  r.x = (unsigned int)f2bf(f.x) | ((unsigned int)f2bf(f.y) << 16);
  r.y = (unsigned int)f2bf(f.z) | ((unsigned int)f2bf(f.w) << 16);
  ((uint2*)dst)[i] = r;
}

// ---------- WMMA GEMM:  Y[M,N] = X[M,Kd] * W[N,Kd]^T  (pre-converted bf16 operands) ----------
// Each wave: one 16-row A fragment shared across NT adjacent 16-col N tiles (4 independent
// WMMA accumulator chains for latency hiding). Out-of-range columns: B row pointer clamped
// to row 0 (garbage stays in unstored lanes' accumulators; D col j depends only on B col j).
// Optional column split: cols [0,nsplit) -> Y0 (ld = nsplit), cols [nsplit,N) -> Y1.
__global__ void k_gemm_bf(const unsigned short* __restrict__ Xb,
                          const unsigned short* __restrict__ Wb,
                          float* __restrict__ Y0, float* __restrict__ Y1,
                          int M, int N, int Kd, int nsplit) {
  const int wave = blockIdx.x * (blockDim.x >> 5) + (threadIdx.x >> 5);
  const int tn   = (N + 15) >> 4;
  const int tng  = (tn + NT - 1) / NT;            // groups of NT tiles
  const int tm   = M >> 4;
  if (wave >= tm * tng) return;                   // wave-uniform exit (EXEC all-1s for WMMA)
  const int tile_m = wave / tng, tile_g = wave % tng;
  const int m0 = tile_m << 4, n0 = tile_g * (NT * 16);
  const int lane  = threadIdx.x & 31;
  const int khalf = (lane >> 4) << 3;             // A: upper half-lanes hold K-sub 8..15 / 24..31
  const int kbase = (lane >> 4) << 3;             // B: upper half-lanes hold K-sub 8..15 / 24..31
  const int rowA  = m0 + (lane & 15);

  const unsigned short* pa = Xb + rowA * Kd;
  const unsigned short* pb[NT];
  int col[NT];
  #pragma unroll
  for (int i = 0; i < NT; ++i) {
    col[i] = n0 + i * 16 + (lane & 15);
    pb[i]  = Wb + (col[i] < N ? col[i] : 0) * Kd + kbase;
  }

  v8f acc[NT];
  const v8f vzero = {};
  #pragma unroll
  for (int i = 0; i < NT; ++i) acc[i] = vzero;

  for (int kk = 0; kk < Kd; kk += 32) {
    union Uv { v16bf v; uint4 q[2]; } Am;
    // A fragment: two contiguous 8-element bf16 chunks (ISA 16-bit A 16x32 layout)
    Am.q[0] = *(const uint4*)(pa + kk + khalf);
    Am.q[1] = *(const uint4*)(pa + kk + 16 + khalf);
    #pragma unroll
    for (int i = 0; i < NT; ++i) {
      union Uv Bm;
      // B fragment: two contiguous 8-element bf16 chunks along K (same K order as A)
      Bm.q[0] = *(const uint4*)(pb[i] + kk);
      Bm.q[1] = *(const uint4*)(pb[i] + kk + 16);
      acc[i] = __builtin_amdgcn_wmma_f32_16x16x32_bf16(false, Am.v, false, Bm.v,
                                                       (short)0, acc[i], false, false);
    }
  }
  // C/D layout: VGPR j -> row m0 + j (+8 for upper half-lanes), col = n + lane%16
  const int rbase = m0 + ((lane >> 4) << 3);
  #pragma unroll
  for (int i = 0; i < NT; ++i) {
    const int c = col[i];
    if (c >= N) continue;
    if (c < nsplit) {
      for (int j = 0; j < 8; ++j) Y0[(rbase + j) * nsplit + c] = acc[i][j];
    } else {
      const int ld1 = N - nsplit;
      for (int j = 0; j < 8; ++j) Y1[(rbase + j) * ld1 + (c - nsplit)] = acc[i][j];
    }
  }
}

// ---------- depthwise 3x3 conv (SAME) + bias + SiLU; emits f32 (scan) + bf16 (GEMM) ----------
__global__ void k_conv_silu(const float* __restrict__ xc, const float* __restrict__ cw,
                            const float* __restrict__ cb, float* __restrict__ u,
                            unsigned short* __restrict__ u_bf) {
  int idx = blockIdx.x * blockDim.x + threadIdx.x;        // T_*D_INNER
  if (idx >= T_ * D_INNER) return;
  int d = idx % D_INNER, tok = idx / D_INNER;
  int b = tok / L_, p = tok % L_, h = p >> 6, w = p & 63;
  float acc = cb[d];
  #pragma unroll
  for (int ky = 0; ky < 3; ++ky) {
    int hh = h + ky - 1;
    if ((unsigned)hh >= H_) continue;
    #pragma unroll
    for (int kx = 0; kx < 3; ++kx) {
      int ww = w + kx - 1;
      if ((unsigned)ww >= W_) continue;
      acc += xc[((b * L_) + (hh << 6) + ww) * D_INNER + d] * cw[d * 9 + ky * 3 + kx];
    }
  }
  float s = silu_f(acc);
  u[idx] = s;
  u_bf[idx] = f2bf(s);
}

// ---------- scan pass 1: per chunk, h0=0; emit (prod(a), h_final) per (b,k,d,n) ----------
__global__ void k_scan_pass1(const float* __restrict__ u, const float* __restrict__ xdbl,
                             const float* __restrict__ dtw, const float* __restrict__ dtb,
                             const float* __restrict__ A_logs,
                             float* __restrict__ Aprod, float* __restrict__ Hfin) {
  int blk = blockIdx.x;                  // B_*KDIR*NCHUNK = 256
  int chunk = blk & (NCHUNK - 1);
  int bk = blk >> 5;
  int k = bk & 3, b = bk >> 2;
  int d = threadIdx.x;                   // 0..191
  float A[D_STATE], wdt[DT_RANK];
  #pragma unroll
  for (int n = 0; n < D_STATE; ++n) A[n] = -__expf(A_logs[(k * D_INNER + d) * D_STATE + n]);
  #pragma unroll
  for (int r = 0; r < DT_RANK; ++r) wdt[r] = dtw[(k * D_INNER + d) * DT_RANK + r];
  float bias = dtb[k * D_INNER + d];
  float h[D_STATE], ap[D_STATE];
  #pragma unroll
  for (int n = 0; n < D_STATE; ++n) { h[n] = 0.f; ap[n] = 1.f; }

  for (int t = 0; t < LCH; ++t) {
    int s = chunk * LCH + t;
    int p = dir_pos(k, s);
    int tok = b * L_ + p;
    const float* xd = xdbl + tok * CPROJ + k * 38;
    float dt = bias;
    #pragma unroll
    for (int r = 0; r < DT_RANK; ++r) dt += wdt[r] * xd[r];
    dt = softplus_f(dt);
    float du = dt * u[tok * D_INNER + d];
    #pragma unroll
    for (int n = 0; n < D_STATE; ++n) {
      float a = __expf(dt * A[n]);
      h[n] = h[n] * a + du * xd[DT_RANK + n];
      ap[n] *= a;
    }
  }
  int seq = ((b * KDIR + k) * D_INNER + d) * D_STATE;
  #pragma unroll
  for (int n = 0; n < D_STATE; ++n) {
    Aprod[(seq + n) * NCHUNK + chunk] = ap[n];
    Hfin [(seq + n) * NCHUNK + chunk] = h[n];
  }
}

// ---------- scan middle: stitch chunk boundaries (24576 scalar recurrences x 32) ----------
__global__ void k_scan_mid(const float* __restrict__ Aprod, const float* __restrict__ Hfin,
                           float* __restrict__ Hinit) {
  int idx = blockIdx.x * blockDim.x + threadIdx.x;   // B_*KDIR*D_INNER*D_STATE = 24576
  if (idx >= B_ * KDIR * D_INNER * D_STATE) return;
  int base = idx * NCHUNK;
  float carry = 0.f;
  #pragma unroll
  for (int c = 0; c < NCHUNK; ++c) {
    Hinit[base + c] = carry;
    carry = Aprod[base + c] * carry + Hfin[base + c];
  }
}

// ---------- scan pass 3: replay with correct h0, emit y + D*u at position p ----------
__global__ void k_scan_pass2(const float* __restrict__ u, const float* __restrict__ xdbl,
                             const float* __restrict__ dtw, const float* __restrict__ dtb,
                             const float* __restrict__ A_logs, const float* __restrict__ Ds,
                             const float* __restrict__ Hinit, float* __restrict__ ys) {
  int blk = blockIdx.x;
  int chunk = blk & (NCHUNK - 1);
  int bk = blk >> 5;
  int k = bk & 3, b = bk >> 2;
  int d = threadIdx.x;
  float A[D_STATE], wdt[DT_RANK];
  #pragma unroll
  for (int n = 0; n < D_STATE; ++n) A[n] = -__expf(A_logs[(k * D_INNER + d) * D_STATE + n]);
  #pragma unroll
  for (int r = 0; r < DT_RANK; ++r) wdt[r] = dtw[(k * D_INNER + d) * DT_RANK + r];
  float bias = dtb[k * D_INNER + d];
  float Dv = Ds[k * D_INNER + d];
  int seq = ((b * KDIR + k) * D_INNER + d) * D_STATE;
  float h[D_STATE];
  #pragma unroll
  for (int n = 0; n < D_STATE; ++n) h[n] = Hinit[(seq + n) * NCHUNK + chunk];

  for (int t = 0; t < LCH; ++t) {
    int s = chunk * LCH + t;
    int p = dir_pos(k, s);
    int tok = b * L_ + p;
    const float* xd = xdbl + tok * CPROJ + k * 38;
    float dt = bias;
    #pragma unroll
    for (int r = 0; r < DT_RANK; ++r) dt += wdt[r] * xd[r];
    dt = softplus_f(dt);
    float uu = u[tok * D_INNER + d];
    float du = dt * uu;
    float y = 0.f;
    #pragma unroll
    for (int n = 0; n < D_STATE; ++n) {
      float a = __expf(dt * A[n]);
      h[n] = h[n] * a + du * xd[DT_RANK + n];
      y += h[n] * xd[DT_RANK + D_STATE + n];
    }
    ys[((b * KDIR + k) * L_ + p) * D_INNER + d] = y + Dv * uu;
  }
}

// ---------- combine 4 directions + LayerNorm(d_inner) + SiLU(z) gating -> bf16 ----------
__global__ void k_ln_gate(const float* __restrict__ ys, const float* __restrict__ z,
                          const float* __restrict__ lnw, const float* __restrict__ lnb,
                          unsigned short* __restrict__ yg_bf) {
  int tok = blockIdx.x;                  // T_
  int d = threadIdx.x;                   // 192 threads = 6 wave32
  int b = tok / L_, p = tok % L_;
  float y = 0.f;
  #pragma unroll
  for (int k = 0; k < KDIR; ++k) y += ys[((b * KDIR + k) * L_ + p) * D_INNER + d];
  float v = y, v2 = y * y;
  #pragma unroll
  for (int off = 16; off > 0; off >>= 1) {
    v  += __shfl_down(v, off, 32);
    v2 += __shfl_down(v2, off, 32);
  }
  __shared__ float s1[6], s2[6];
  __shared__ float mu, rstd;
  if ((threadIdx.x & 31) == 0) { s1[threadIdx.x >> 5] = v; s2[threadIdx.x >> 5] = v2; }
  __syncthreads();
  if (threadIdx.x == 0) {
    float S = 0.f, S2 = 0.f;
    #pragma unroll
    for (int i = 0; i < 6; ++i) { S += s1[i]; S2 += s2[i]; }
    float m = S / (float)D_INNER;
    float var = S2 / (float)D_INNER - m * m;
    mu = m; rstd = rsqrtf(var + 1e-5f);
  }
  __syncthreads();
  float yn = (y - mu) * rstd * lnw[d] + lnb[d];
  float zz = z[tok * D_INNER + d];
  yg_bf[tok * D_INNER + d] = f2bf(yn * silu_f(zz));
}

// ---------- host ----------
static inline void launch_gemm(const unsigned short* Xb, const unsigned short* Wb,
                               float* Y0, float* Y1, int M, int N, int Kd, int nsplit,
                               hipStream_t s) {
  int tng = (((N + 15) >> 4) + NT - 1) / NT;
  int waves = (M >> 4) * tng;
  int blocks = (waves + 3) / 4;
  k_gemm_bf<<<blocks, 128, 0, s>>>(Xb, Wb, Y0, Y1, M, N, Kd, nsplit);
}
static inline void launch_cvt(const float* src, unsigned short* dst, int n, hipStream_t s) {
  int n4 = n >> 2;
  k_cvt_bf16<<<(n4 + 255) / 256, 256, 0, s>>>(src, dst, n4);
}

extern "C" void kernel_launch(void* const* d_in, const int* in_sizes, int n_in,
                              void* d_out, int out_size, void* d_ws, size_t ws_size,
                              hipStream_t stream) {
  (void)in_sizes; (void)n_in; (void)out_size; (void)ws_size;
  const float* x        = (const float*)d_in[0];
  const float* in_proj  = (const float*)d_in[1];
  const float* conv_w   = (const float*)d_in[2];
  const float* conv_b   = (const float*)d_in[3];
  const float* x_proj_w = (const float*)d_in[4];   // (4,38,192) contiguous = 152 x 192
  const float* dt_w     = (const float*)d_in[5];   // (4,192,6)
  const float* dt_b     = (const float*)d_in[6];   // (4,192)
  const float* A_logs   = (const float*)d_in[7];   // (768,16)
  const float* Ds       = (const float*)d_in[8];   // (768,)
  const float* ln_w     = (const float*)d_in[9];
  const float* ln_b     = (const float*)d_in[10];
  const float* out_w    = (const float*)d_in[11];  // (96,192)
  float* out = (float*)d_out;
  float* ws  = (float*)d_ws;

  // f32 region
  float* xc    = ws;                                    // T_*192
  float* z     = xc    + (size_t)T_ * D_INNER;
  float* u     = z     + (size_t)T_ * D_INNER;
  float* xdbl  = u     + (size_t)T_ * D_INNER;          // T_*152
  float* Aprod = xdbl  + (size_t)T_ * CPROJ;            // 24576*32
  float* Hfin  = Aprod + (size_t)24576 * NCHUNK;
  float* Hinit = Hfin  + (size_t)24576 * NCHUNK;
  float* ys    = Hinit + (size_t)24576 * NCHUNK;        // B*K*L*192
  // bf16 region (16B-aligned: f32 region size is a multiple of 16 bytes)
  unsigned short* x_bf   = (unsigned short*)(ys + (size_t)B_ * KDIR * L_ * D_INNER);
  unsigned short* wi_bf  = x_bf  + (size_t)T_ * D_MODEL;        // 384*96
  unsigned short* u_bf   = wi_bf + (size_t)384 * 96;            // T_*192
  unsigned short* wx_bf  = u_bf  + (size_t)T_ * D_INNER;        // 152*192
  unsigned short* wo_bf  = wx_bf + (size_t)CPROJ * D_INNER;     // 96*192
  unsigned short* yg_bf  = wo_bf + (size_t)D_MODEL * D_INNER;   // T_*192

  // 0) one-time bf16 conversions (inputs + weights)
  launch_cvt(x,        x_bf,  T_ * D_MODEL,      stream);
  launch_cvt(in_proj,  wi_bf, 384 * 96,          stream);
  launch_cvt(x_proj_w, wx_bf, CPROJ * D_INNER,   stream);
  launch_cvt(out_w,    wo_bf, D_MODEL * D_INNER, stream);
  // 1) in_proj GEMM: (8192x96)x(96x384)^T -> split xc | z
  launch_gemm(x_bf, wi_bf, xc, z, T_, 2 * D_INNER, D_MODEL, D_INNER, stream);
  // 2) depthwise conv + SiLU (emits f32 + bf16)
  k_conv_silu<<<(T_ * D_INNER + 255) / 256, 256, 0, stream>>>(xc, conv_w, conv_b, u, u_bf);
  // 3) x_proj GEMM: (8192x192)x(192x152)^T -> xdbl (per-token, all 4 directions)
  launch_gemm(u_bf, wx_bf, xdbl, nullptr, T_, CPROJ, D_INNER, CPROJ, stream);
  // 4-6) chunked selective scan (3 passes)
  k_scan_pass1<<<B_ * KDIR * NCHUNK, D_INNER, 0, stream>>>(u, xdbl, dt_w, dt_b, A_logs,
                                                           Aprod, Hfin);
  k_scan_mid<<<(24576 + 255) / 256, 256, 0, stream>>>(Aprod, Hfin, Hinit);
  k_scan_pass2<<<B_ * KDIR * NCHUNK, D_INNER, 0, stream>>>(u, xdbl, dt_w, dt_b, A_logs,
                                                           Ds, Hinit, ys);
  // 7) combine directions + LayerNorm + gate -> bf16
  k_ln_gate<<<T_, D_INNER, 0, stream>>>(ys, z, ln_w, ln_b, yg_bf);
  // 8) out_proj GEMM: (8192x192)x(192x96)^T -> out
  launch_gemm(yg_bf, wo_bf, out, nullptr, T_, D_MODEL, D_INNER, D_MODEL, stream);
}
